// FocusedLinearAttention_13099650253008
// MI455X (gfx1250) — matrix operand order repaired and checked by reading, not compiled, not run on hardware
//
#include <hip/hip_runtime.h>

#define BATCH 1024
#define NTOK  256
#define DIMC  384
#define HEADS 12
#define HD    32
#define QKVC  1152
#define BN    (BATCH * NTOK)

typedef _Float16 v16h __attribute__((ext_vector_type(16)));
typedef _Float16 v8h  __attribute__((ext_vector_type(8)));
typedef float    v8f  __attribute__((ext_vector_type(8)));

// A-fragment (16x32 f16, MxK): lane (hi, m=lane&15) holds halves
//   j=0..15 -> K = k0 + hi*16 + j
// rowptr points at element [m][0] of a K-contiguous row.
__device__ inline v16h load_a_frag(const _Float16* rowptr, int k0, int hi) {
    v8h lo = *(const v8h*)(rowptr + k0 + hi * 16);
    v8h up = *(const v8h*)(rowptr + k0 + hi * 16 + 8);
    v16h a;
#pragma unroll
    for (int i = 0; i < 8; ++i) { a[i] = lo[i]; a[8 + i] = up[i]; }
    return a;
}

// ---------------------------------------------------------------------------
// K0: weight conversion + inverse softplus scale
// ---------------------------------------------------------------------------
__global__ void prep_kernel(const float* __restrict__ w_qkv,
                            const float* __restrict__ w_proj,
                            const float* __restrict__ scale_p,
                            _Float16* __restrict__ wq16,
                            _Float16* __restrict__ wp16,
                            float* __restrict__ inv_scale) {
    int i = blockIdx.x * blockDim.x + threadIdx.x;
    const int n1 = QKVC * DIMC, n2 = DIMC * DIMC;
    if (i < n1) {
        wq16[i] = (_Float16)w_qkv[i];
    } else if (i < n1 + n2) {
        wp16[i - n1] = (_Float16)w_proj[i - n1];
    } else if (i < n1 + n2 + DIMC) {
        int c = i - n1 - n2;
        float sp = log1pf(expf(scale_p[c]));   // softplus
        inv_scale[c] = 1.0f / sp;
    }
}

// ---------------------------------------------------------------------------
// Row-wise focus renorm for a 16-token slab staged in LDS `work`:
//   dst_row = row^3 * (||row|| / ||row^3||)
// ---------------------------------------------------------------------------
__device__ inline void focus_rows(float (*work)[388], float* rowfac,
                                  _Float16* __restrict__ dst, long row0, int tid) {
    const int wave = tid >> 5, lane = tid & 31;
#pragma unroll
    for (int rr = 0; rr < 2; ++rr) {
        int m = wave + 8 * rr;
        float s2 = 0.f, s6 = 0.f;
        for (int c = lane; c < DIMC; c += 32) {
            float v = work[m][c];
            float v2 = v * v;
            s2 += v2;
            s6 += v2 * v2 * v2;       // (v^3)^2
        }
        for (int off = 16; off > 0; off >>= 1) {
            s2 += __shfl_down(s2, off, 32);
            s6 += __shfl_down(s6, off, 32);
        }
        if (lane == 0) rowfac[m] = sqrtf(s2) / sqrtf(s6);   // ||v|| / ||v^3||
    }
    __syncthreads();
    for (int idx = tid; idx < 16 * DIMC; idx += 256) {
        int m = idx / DIMC, c = idx % DIMC;
        float v = work[m][c];
        dst[(row0 + m) * DIMC + c] = (_Float16)(v * v * v * rowfac[m]);
    }
}

// ---------------------------------------------------------------------------
// K1: fused QKV GEMM + bias + pos_enc + ReLU + scale + focus renorm -> f16
// One block = 32 tokens (two 16-row WMMA sub-tiles sharing each B fragment:
// halves the L2 weight-fragment traffic vs a 16-row tile). The 384-channel
// norms are done in LDS per 16-row half; f32 qkv never touches HBM.
// ---------------------------------------------------------------------------
__global__ __launch_bounds__(256) void qkv_focus_kernel(
    const float* __restrict__ x, const _Float16* __restrict__ wq16,
    const float* __restrict__ b_qkv, const float* __restrict__ pos_enc,
    const float* __restrict__ inv_scale,
    _Float16* __restrict__ qf, _Float16* __restrict__ kf,
    _Float16* __restrict__ vf) {
    __shared__ alignas(32) _Float16 Atile[32][392];   // 32 tokens x 384 (pad 8)
    __shared__ float work[16][388];
    __shared__ float rowfac[16];

    const int tid = threadIdx.x;
    const int wave = tid >> 5, lane = tid & 31;
    const int hi = lane >> 4, lrow = lane & 15;
    const long row0 = (long)blockIdx.x * 32;
    const int nbase = (int)(row0 & (NTOK - 1));       // token index within batch

    // stage x -> LDS f16 (vectorized float4 loads, 4-half stores)
#pragma unroll
    for (int i = 0; i < 12; ++i) {
        int idx = tid + 256 * i;                      // 32*96 float4 chunks
        int m = idx / 96, c4 = idx % 96;
        const float4 f = *(const float4*)(x + (row0 + m) * DIMC + c4 * 4);
        _Float16* dst = &Atile[m][c4 * 4];
        dst[0] = (_Float16)f.x; dst[1] = (_Float16)f.y;
        dst[2] = (_Float16)f.z; dst[3] = (_Float16)f.w;
    }
    __syncthreads();

    v8f acc[2][9] = {};                               // 2 M-subtiles x 9 col-tiles
    for (int kc = 0; kc < 12; ++kc) {
        const int k0 = kc * 32;
        v16h a0 = load_a_frag(&Atile[lrow][0], k0, hi);
        v16h a1 = load_a_frag(&Atile[16 + lrow][0], k0, hi);
#pragma unroll
        for (int t = 0; t < 9; ++t) {
            int col = (wave + 8 * t) * 16 + lrow;
            // B[k][col] = w_qkv[col][k] : 16 contiguous K halves per lane
            v16h b = *(const v16h*)(wq16 + (long)col * DIMC + k0 + hi * 16);
            acc[0][t] = __builtin_amdgcn_wmma_f32_16x16x32_f16(
                false, a0, false, b, (short)0, acc[0][t], false, false);
            acc[1][t] = __builtin_amdgcn_wmma_f32_16x16x32_f16(
                false, a1, false, b, (short)0, acc[1][t], false, false);
        }
    }

    // ---- V epilogue: bias only, write f16 (both 16-row halves) ----
#pragma unroll
    for (int t = 0; t < 9; ++t) {
        int col = (wave + 8 * t) * 16 + lrow;
        if (col >= 2 * DIMC) {
            int c = col - 2 * DIMC;
            float bias = b_qkv[col];
#pragma unroll
            for (int s = 0; s < 2; ++s)
#pragma unroll
                for (int r = 0; r < 8; ++r) {
                    int m = s * 16 + hi * 8 + r;
                    vf[(row0 + m) * DIMC + c] = (_Float16)(acc[s][t][r] + bias);
                }
        }
    }

    // ---- Q phase: bias, relu+eps, /softplus(scale), focus renorm ----
    for (int s = 0; s < 2; ++s) {
#pragma unroll
        for (int t = 0; t < 9; ++t) {
            int col = (wave + 8 * t) * 16 + lrow;
            if (col < DIMC) {
                float bias = b_qkv[col];
                float isc = inv_scale[col];
#pragma unroll
                for (int r = 0; r < 8; ++r) {
                    float v = acc[s][t][r] + bias;
                    v = fmaxf(v, 0.f) + 1e-6f;
                    work[hi * 8 + r][col] = v * isc;
                }
            }
        }
        __syncthreads();
        focus_rows(work, rowfac, qf, row0 + s * 16, tid);
        __syncthreads();
    }

    // ---- K phase: bias + pos_enc, relu+eps, /scale, focus renorm ----
    for (int s = 0; s < 2; ++s) {
#pragma unroll
        for (int t = 0; t < 9; ++t) {
            int col = (wave + 8 * t) * 16 + lrow;
            if (col >= DIMC && col < 2 * DIMC) {
                int c = col - DIMC;
                float bias = b_qkv[col];
                float isc = inv_scale[c];
#pragma unroll
                for (int r = 0; r < 8; ++r) {
                    int m16 = hi * 8 + r;
                    float v = acc[s][t][r] + bias +
                              pos_enc[(nbase + s * 16 + m16) * DIMC + c];
                    v = fmaxf(v, 0.f) + 1e-6f;
                    work[m16][c] = v * isc;
                }
            }
        }
        __syncthreads();
        focus_rows(work, rowfac, kf, row0 + s * 16, tid);
        __syncthreads();
    }
}

// ---------------------------------------------------------------------------
// K2: fused per-(b,h) linear attention: kv = k^T v / N (WMMA),
//     z = 1/(q . mean(k) + eps), out = (q @ kv) * z   -> attn f32 (d_out scratch)
// ---------------------------------------------------------------------------
__global__ __launch_bounds__(256) void attn_kernel(
    const _Float16* __restrict__ qf, const _Float16* __restrict__ kf,
    const _Float16* __restrict__ vf, float* __restrict__ attn_out) {
    __shared__ alignas(32) _Float16 kT[32][272];   // [d][n]
    __shared__ alignas(32) _Float16 vT[32][272];   // [e][n]
    __shared__ alignas(32) _Float16 kvT[32][48];   // [e][d]  (B operand: K=d contiguous)
    __shared__ float kmean[32];
    __shared__ float zbuf[256];

    const int tid = threadIdx.x;
    const int wave = tid >> 5, lane = tid & 31;
    const int hi = lane >> 4, lrow = lane & 15;
    const int b = blockIdx.x / HEADS, h = blockIdx.x % HEADS;
    const long base = ((long)b * NTOK) * DIMC + h * HD;

    {   // stage k,v transposed into LDS
        const _Float16* kp = kf + base + (long)tid * DIMC;
        const _Float16* vp = vf + base + (long)tid * DIMC;
#pragma unroll
        for (int e = 0; e < HD; ++e) { kT[e][tid] = kp[e]; vT[e][tid] = vp[e]; }
    }
    __syncthreads();

    if (wave < 4) {     // kv = k^T v / N, 2x2 tiles of 16x16, K=256
        const int ti = wave >> 1, tj = wave & 1;
        v8f acc = {};
        for (int k0 = 0; k0 < NTOK; k0 += 32) {
            v16h a = load_a_frag(&kT[ti * 16 + lrow][0], k0, hi);
            v16h bb = *(const v16h*)(&vT[tj * 16 + lrow][k0 + hi * 16]);
            acc = __builtin_amdgcn_wmma_f32_16x16x32_f16(
                false, a, false, bb, (short)0, acc, false, false);
        }
        const float invN = 1.0f / (float)NTOK;
#pragma unroll
        for (int r = 0; r < 8; ++r) {
            int d = ti * 16 + hi * 8 + r;
            int e = tj * 16 + lrow;
            kvT[e][d] = (_Float16)(acc[r] * invN);
        }
    } else if (wave == 4) {   // k_mean over tokens
        float s = 0.f;
        for (int n = 0; n < NTOK; ++n) s += (float)kT[lane][n];
        kmean[lane] = s * (1.0f / (float)NTOK);
    }
    __syncthreads();

    {   // z[n] = 1 / (q[n] . k_mean + eps)
        const _Float16* qp = qf + base + (long)tid * DIMC;
        float s = 0.f;
#pragma unroll
        for (int e = 0; e < HD; ++e) s += (float)qp[e] * kmean[e];
        zbuf[tid] = 1.0f / (s + 1e-6f);
    }
    __syncthreads();

    // out = q @ kv : each wave owns M-tiles {2w,2w+1} x N-tiles {0,1}, K=32
#pragma unroll
    for (int mt2 = 0; mt2 < 2; ++mt2) {
        const int mt = wave * 2 + mt2;
        const _Float16* qrow = qf + base + (long)(mt * 16 + lrow) * DIMC;
        v16h a = load_a_frag(qrow, 0, hi);
#pragma unroll
        for (int tj = 0; tj < 2; ++tj) {
            v16h bb = *(const v16h*)(&kvT[tj * 16 + lrow][hi * 16]);
            v8f acc = {};
            acc = __builtin_amdgcn_wmma_f32_16x16x32_f16(
                false, a, false, bb, (short)0, acc, false, false);
#pragma unroll
            for (int r = 0; r < 8; ++r) {
                int n = mt * 16 + hi * 8 + r;
                int e = tj * 16 + lrow;
                attn_out[base + (long)n * DIMC + e] = acc[r] * zbuf[n];
            }
        }
    }
}

// ---------------------------------------------------------------------------
// K3: 5x5 depthwise conv on v (per head image 16x16xd) + attn residual -> y f16
// ---------------------------------------------------------------------------
__global__ __launch_bounds__(256) void dwc_kernel(
    const _Float16* __restrict__ vf, const float* __restrict__ w_dwc,
    const float* __restrict__ b_dwc, const float* __restrict__ attn_in,
    _Float16* __restrict__ yf) {
    __shared__ _Float16 vimg[256][33];   // [pixel][channel]
    __shared__ float wsm[HD * 25];
    __shared__ float bsm[HD];

    const int tid = threadIdx.x;
    const int b = blockIdx.x / HEADS, h = blockIdx.x % HEADS;
    const long base = ((long)b * NTOK) * DIMC + h * HD;

    {   const _Float16* vp = vf + base + (long)tid * DIMC;
#pragma unroll
        for (int dd = 0; dd < HD; ++dd) vimg[tid][dd] = vp[dd];
    }
    for (int i = tid; i < HD * 25; i += 256) wsm[i] = w_dwc[i];
    if (tid < HD) bsm[tid] = b_dwc[tid];
    __syncthreads();

    const int py = tid >> 4, px = tid & 15;
    for (int dd = 0; dd < HD; ++dd) {
        float s = bsm[dd];
#pragma unroll
        for (int ky = -2; ky <= 2; ++ky) {
            int yy = py + ky;
            if (yy < 0 || yy > 15) continue;
#pragma unroll
            for (int kx = -2; kx <= 2; ++kx) {
                int xx = px + kx;
                if (xx < 0 || xx > 15) continue;
                s += (float)vimg[yy * 16 + xx][dd] *
                     wsm[dd * 25 + (ky + 2) * 5 + (kx + 2)];
            }
        }
        float o = s + attn_in[base + (long)tid * DIMC + dd];
        yf[base + (long)tid * DIMC + dd] = (_Float16)o;
    }
}

// ---------------------------------------------------------------------------
// K4: output projection y @ w_proj^T + b_proj  (WMMA) -> d_out f32
// A tile (pure f16 copy) is staged with GLOBAL_LOAD_ASYNC_TO_LDS_B128
// (ASYNCcnt-tracked, per-lane async copy; no VGPR round trip).
// ---------------------------------------------------------------------------
__global__ __launch_bounds__(256) void proj_kernel(
    const _Float16* __restrict__ yf, const _Float16* __restrict__ wp16,
    const float* __restrict__ b_proj, float* __restrict__ out) {
    __shared__ alignas(32) _Float16 Atile[32][392];   // sole LDS object -> offset 0
    const int tid = threadIdx.x;
    const int wave = tid >> 5, lane = tid & 31;
    const int hi = lane >> 4, lrow = lane & 15;
    const long row0 = (long)blockIdx.x * 32;

    // async-copy 32 rows x 768B (48 x 16B chunks per row) into LDS
#pragma unroll
    for (int i = 0; i < 6; ++i) {
        int idx = tid + 256 * i;                 // 1536 chunks total
        int m = idx / 48, c = idx % 48;
        unsigned lds_off = (unsigned)(m * (392 * 2) + c * 16);
        const _Float16* g = yf + (row0 + m) * DIMC + c * 8;
        asm volatile("global_load_async_to_lds_b128 %0, %1, off"
                     :: "v"(lds_off), "v"(g) : "memory");
    }
    asm volatile("s_wait_asynccnt 0x0" ::: "memory");
    __syncthreads();

    v8f acc[2][3] = {};                          // 2 M-subtiles x 3 col-tiles
    for (int kc = 0; kc < 12; ++kc) {
        const int k0 = kc * 32;
        v16h a0 = load_a_frag(&Atile[lrow][0], k0, hi);
        v16h a1 = load_a_frag(&Atile[16 + lrow][0], k0, hi);
#pragma unroll
        for (int t = 0; t < 3; ++t) {
            int col = (wave + 8 * t) * 16 + lrow;
            v16h b = *(const v16h*)(wp16 + (long)col * DIMC + k0 + hi * 16);
            acc[0][t] = __builtin_amdgcn_wmma_f32_16x16x32_f16(
                false, a0, false, b, (short)0, acc[0][t], false, false);
            acc[1][t] = __builtin_amdgcn_wmma_f32_16x16x32_f16(
                false, a1, false, b, (short)0, acc[1][t], false, false);
        }
    }
#pragma unroll
    for (int t = 0; t < 3; ++t) {
        int col = (wave + 8 * t) * 16 + lrow;
        float bias = b_proj[col];
#pragma unroll
        for (int s = 0; s < 2; ++s)
#pragma unroll
            for (int r = 0; r < 8; ++r) {
                int m = s * 16 + hi * 8 + r;
                out[(row0 + m) * DIMC + col] = acc[s][t][r] + bias;
            }
    }
}

// ---------------------------------------------------------------------------
extern "C" void kernel_launch(void* const* d_in, const int* in_sizes, int n_in,
                              void* d_out, int out_size, void* d_ws, size_t ws_size,
                              hipStream_t stream) {
    (void)in_sizes; (void)n_in; (void)out_size; (void)ws_size;
    const float* x       = (const float*)d_in[0];
    const float* w_qkv   = (const float*)d_in[1];
    const float* b_qkv   = (const float*)d_in[2];
    const float* w_proj  = (const float*)d_in[3];
    const float* b_proj  = (const float*)d_in[4];
    const float* scale_p = (const float*)d_in[5];
    const float* pos_enc = (const float*)d_in[6];
    const float* w_dwc   = (const float*)d_in[7];
    const float* b_dwc   = (const float*)d_in[8];
    float* out = (float*)d_out;

    char* ws = (char*)d_ws;
    size_t off = 0;
    auto wsalloc = [&](size_t bytes) -> void* {
        void* p = ws + off;
        off += (bytes + 255) & ~(size_t)255;
        return p;
    };
    _Float16* wq16      = (_Float16*)wsalloc((size_t)QKVC * DIMC * 2);
    _Float16* wp16      = (_Float16*)wsalloc((size_t)DIMC * DIMC * 2);
    float*    inv_scale = (float*)   wsalloc((size_t)DIMC * 4);
    _Float16* qf        = (_Float16*)wsalloc((size_t)BN * DIMC * 2);
    _Float16* kf        = (_Float16*)wsalloc((size_t)BN * DIMC * 2);
    _Float16* vf        = (_Float16*)wsalloc((size_t)BN * DIMC * 2);
    _Float16* yf        = (_Float16*)wsalloc((size_t)BN * DIMC * 2);

    const int prep_n = QKVC * DIMC + DIMC * DIMC + DIMC;
    prep_kernel<<<(prep_n + 255) / 256, 256, 0, stream>>>(
        w_qkv, w_proj, scale_p, wq16, wp16, inv_scale);

    qkv_focus_kernel<<<BN / 32, 256, 0, stream>>>(
        x, wq16, b_qkv, pos_enc, inv_scale, qf, kf, vf);

    // attn written into d_out (scratch), later overwritten by proj_kernel
    attn_kernel<<<BATCH * HEADS, 256, 0, stream>>>(qf, kf, vf, out);

    dwc_kernel<<<BATCH * HEADS, 256, 0, stream>>>(vf, w_dwc, b_dwc, out, yf);

    proj_kernel<<<BN / 32, 256, 0, stream>>>(yf, wp16, b_proj, out);
}
